// VRNN_INDEP_4294967296020
// MI455X (gfx1250) — matrix operand order repaired and checked by reading, not compiled, not run on hardware
//
#include <hip/hip_runtime.h>
#include <math.h>

// ---------------- problem constants ----------------
constexpr int kT = 128, kB = 512, kA = 5;
constexpr int kX = 2, kY = 10, kZ = 32, kH = 128, kR = 128;
constexpr float kLOG2PI = 1.8378770664093453f;

// ---------------- WMMA vector types ----------------
typedef __attribute__((ext_vector_type(16))) __bf16 v16bf;
typedef __attribute__((ext_vector_type(8)))  __bf16 v8bf;
typedef __attribute__((ext_vector_type(8)))  float  v8f;

// ---------------- bf16 weight workspace layout (elements) ----------------
// Matrix-major, each matrix holds kA consecutive per-agent blocks of Kpad*N.
constexpr size_t W_ENC_W1  = 0;            // 160x128 per a
constexpr size_t W_ENC_W2  = 102400;       // 128x128
constexpr size_t W_ENC_MW  = 184320;       // 128x32
constexpr size_t W_ENC_SW  = 204800;       // 128x32
constexpr size_t W_PRI_W1  = 225280;       // 160x128
constexpr size_t W_PRI_W2  = 327680;       // 128x128
constexpr size_t W_PRI_MW  = 409600;       // 128x32
constexpr size_t W_PRI_SW  = 430080;       // 128x32
constexpr size_t W_DEC_W1  = 450560;       // 192x128
constexpr size_t W_DEC_W2  = 573440;       // 128x128
constexpr size_t W_GRU_WIH = 655360;       // 64x384
constexpr size_t W_GRU_WHH = 778240;       // 128x384
constexpr size_t W_TOTAL   = 1024000;      // elements (~2.0 MB bf16)

// ---------------- prep: f32 [a][K][N] -> bf16 WMMA-B-fragment order ----------------
// Fragment order: flat = (((ntile*nK + ktile)*32 + lane)*16 + e)
// with lane = ((k%16)>=8)*16 + (n%16), e = (k%8) + ((k%32)>=16)*8.  K zero-padded to Kpad.
__global__ void prep_weights(const float* __restrict__ src, __bf16* __restrict__ dst,
                             int K, int Kpad, int N) {
  int total = kA * Kpad * N;
  int idx = blockIdx.x * blockDim.x + threadIdx.x;
  if (idx >= total) return;
  int per = Kpad * N;
  int a = idx / per;
  int d = idx - a * per;
  int e    = d & 15;
  int lane = (d >> 4) & 31;
  int ft   = d >> 9;
  int nK   = Kpad >> 5;
  int kt   = ft % nK;
  int nt   = ft / nK;
  int k = kt * 32 + ((lane >= 16) ? 8 : 0) + (e & 7) + ((e >> 3) << 4);
  int n = nt * 16 + (lane & 15);
  float v = (k < K) ? src[(size_t)(a * K + k) * N + n] : 0.0f;
  dst[idx] = (__bf16)v;
}

__global__ void zero_out_kernel(float* out) {
  if (threadIdx.x < 2) out[threadIdx.x] = 0.0f;
}

// ---------------- WMMA helpers ----------------
// A fragment (16Mx32K bf16) from LDS row-major [16][stride]:
// lane m=lane&15, half=lane>>4; elems 0..7 <- K=[kb+half*8, +8), 8..15 <- K=[kb+16+half*8, +8)
__device__ __forceinline__ v16bf a_frag(const __bf16* row, int kbase, int lhalf) {
  union { v16bf v; v8bf h[2]; } u;
  u.h[0] = *(const v8bf*)(row + kbase + lhalf * 8);
  u.h[1] = *(const v8bf*)(row + kbase + 16 + lhalf * 8);
  return u.v;
}

template <int NK>
__device__ __forceinline__ void gemm_acc(v8f& acc, const __bf16* s_in, int stride,
                                         const __bf16* w, int ntile, int lane) {
  const int m = lane & 15, lhalf = lane >> 4;
  const __bf16* row = s_in + m * stride;
  const v16bf* wp = (const v16bf*)w + (size_t)(ntile * NK) * 32 + lane;
#pragma unroll
  for (int kt = 0; kt < NK; ++kt) {
    v16bf av = a_frag(row, kt * 32, lhalf);
    v16bf bv = wp[(size_t)kt * 32];
    acc = __builtin_amdgcn_wmma_f32_16x16x32_bf16(false, av, false, bv,
                                                  (short)0, acc, false, false);
  }
}

// C/D layout: lanes 0-15: N=lane, M=r; lanes 16-31: N=lane-16, M=8+r
__device__ __forceinline__ void store_relu(__bf16* s_out, int stride, v8f acc,
                                           float bias, int ntile, int lane) {
  const int col = ntile * 16 + (lane & 15);
  const int lhalf = lane >> 4;
#pragma unroll
  for (int r = 0; r < 8; ++r) {
    float v = acc[r] + bias;
    v = v > 0.f ? v : 0.f;
    s_out[(r + 8 * lhalf) * stride + col] = (__bf16)v;
  }
}

__device__ __forceinline__ float softplus_f(float x) {
  return (x > 20.f) ? x : log1pf(__expf(x));
}
__device__ __forceinline__ float sigmoid_f(float x) {
  return 1.f / (1.f + __expf(-x));
}

// ---------------- kernel params ----------------
struct P {
  const float *states, *eps;
  const float *enc_b1, *enc_b2, *enc_mb, *enc_sb;
  const float *pri_b1, *pri_b2, *pri_mb, *pri_sb;
  const float *dec_b1, *dec_b2, *dec_mw, *dec_mb, *dec_sw, *dec_sb;
  const float *gru_bih, *gru_bhh;
  const __bf16 *ws;
  float *out;
};

// ---------------- persistent VRNN kernel ----------------
// grid: kA * (kB/16) = 160 workgroups, 256 threads (8 wave32s).
// Each workgroup owns agent a and a 16-row batch tile for the whole scan.
__global__ __launch_bounds__(256) void vrnn_kernel(P p) {
  __shared__ __bf16 s_xc[16 * 192];   // concat input staging (bf16, K-padded)
  __shared__ __bf16 s_h1[16 * 128];   // hidden1 (bf16)
  __shared__ __bf16 s_h2[16 * 128];   // hidden2 (bf16)
  __shared__ float  s_me[16 * 32];    // enc mean
  __shared__ float  s_se[16 * 32];    // enc std (softplus applied)
  __shared__ float  s_mp[16 * 32];    // pri mean
  __shared__ float  s_sp[16 * 32];    // pri std
  __shared__ __bf16 s_z [16 * 32];    // sampled latent
  __shared__ float  s_hf[16 * 128];   // GRU state f32
  __shared__ __bf16 s_hb[16 * 128];   // GRU state bf16 (WMMA A operand)
  __shared__ float  s_red[16];

  const int tid  = threadIdx.x;
  const int wv   = tid >> 5;     // wave id 0..7 (wave-uniform)
  const int lane = tid & 31;
  const int a    = blockIdx.x >> 5;          // 0..4
  const int b0   = (blockIdx.x & 31) * 16;   // batch row base

  // per-agent bf16 weight blocks (prepped fragment order)
  const __bf16* Wenc1 = p.ws + W_ENC_W1  + (size_t)a * 20480;
  const __bf16* Wenc2 = p.ws + W_ENC_W2  + (size_t)a * 16384;
  const __bf16* Wencm = p.ws + W_ENC_MW  + (size_t)a * 4096;
  const __bf16* Wencs = p.ws + W_ENC_SW  + (size_t)a * 4096;
  const __bf16* Wpri1 = p.ws + W_PRI_W1  + (size_t)a * 20480;
  const __bf16* Wpri2 = p.ws + W_PRI_W2  + (size_t)a * 16384;
  const __bf16* Wprim = p.ws + W_PRI_MW  + (size_t)a * 4096;
  const __bf16* Wpris = p.ws + W_PRI_SW  + (size_t)a * 4096;
  const __bf16* Wdec1 = p.ws + W_DEC_W1  + (size_t)a * 24576;
  const __bf16* Wdec2 = p.ws + W_DEC_W2  + (size_t)a * 16384;
  const __bf16* Wgih  = p.ws + W_GRU_WIH + (size_t)a * 24576;
  const __bf16* Wghh  = p.ws + W_GRU_WHH + (size_t)a * 49152;

  // hoisted per-lane biases (constant over t)
  const int nc   = lane & 15;
  const int colH = wv * 16 + nc;     // column in [0,128)
  const float b_e1 = p.enc_b1[a * kH + colH];
  const float b_e2 = p.enc_b2[a * kH + colH];
  const float b_p1 = p.pri_b1[a * kH + colH];
  const float b_p2 = p.pri_b2[a * kH + colH];
  const float b_d1 = p.dec_b1[a * kH + colH];
  const float b_d2 = p.dec_b2[a * kH + colH];
  float hb_e = 0.f, hb_p = 0.f;
  if (wv < 4) {
    int colz = (wv & 1) * 16 + nc;
    hb_e = (wv < 2) ? p.enc_mb[a * kZ + colz] : p.enc_sb[a * kZ + colz];
    hb_p = (wv < 2) ? p.pri_mb[a * kZ + colz] : p.pri_sb[a * kZ + colz];
  }
  const float b_ir = p.gru_bih[a * 384 + colH];
  const float b_iz = p.gru_bih[a * 384 + 128 + colH];
  const float b_in = p.gru_bih[a * 384 + 256 + colH];
  const float b_hr = p.gru_bhh[a * 384 + colH];
  const float b_hz = p.gru_bhh[a * 384 + 128 + colH];
  const float b_hn = p.gru_bhh[a * 384 + 256 + colH];

  for (int i = tid; i < 16 * 128; i += 256) { s_hf[i] = 0.f; s_hb[i] = (__bf16)0.f; }
  float kl_acc = 0.f, nll_acc = 0.f;
  __syncthreads();

  for (int t = 0; t < kT - 1; ++t) {
    const float* yrow = p.states + (size_t)t * kB * kY;
    const float* xrow = p.states + (size_t)(t + 1) * kB * kY;  // x[r][xi]=xrow[(b0+r)*kY + a*2 + xi]

    // ---- enc input: [x(2) | y(10) | h(128) | pad->160], stride 192 ----
    for (int i = tid; i < 16 * 160; i += 256) {
      int r = i / 160, c = i - r * 160;
      float v;
      if (c < 2)        v = xrow[(b0 + r) * kY + a * 2 + c];
      else if (c < 12)  v = yrow[(b0 + r) * kY + (c - 2)];
      else if (c < 140) v = s_hf[r * 128 + (c - 12)];
      else              v = 0.f;
      s_xc[r * 192 + c] = (__bf16)v;
    }
    __syncthreads();
    { v8f acc = {}; gemm_acc<5>(acc, s_xc, 192, Wenc1, wv, lane); store_relu(s_h1, 128, acc, b_e1, wv, lane); }
    __syncthreads();
    { v8f acc = {}; gemm_acc<4>(acc, s_h1, 128, Wenc2, wv, lane); store_relu(s_h2, 128, acc, b_e2, wv, lane); }
    __syncthreads();
    if (wv < 4) {   // wave-uniform: EXEC all-1s inside, WMMA legal
      v8f acc = {};
      gemm_acc<4>(acc, s_h2, 128, (wv < 2) ? Wencm : Wencs, wv & 1, lane);
      float* dst = (wv < 2) ? s_me : s_se;
      const int colz = (wv & 1) * 16 + nc, lhalf = lane >> 4;
#pragma unroll
      for (int r = 0; r < 8; ++r) {
        float v = acc[r] + hb_e;
        if (wv >= 2) v = softplus_f(v);
        dst[(r + 8 * lhalf) * 32 + colz] = v;
      }
    }
    __syncthreads();

    // ---- pri input: [y(10) | h(128) | pad->160] ----
    for (int i = tid; i < 16 * 160; i += 256) {
      int r = i / 160, c = i - r * 160;
      float v;
      if (c < 10)       v = yrow[(b0 + r) * kY + c];
      else if (c < 138) v = s_hf[r * 128 + (c - 10)];
      else              v = 0.f;
      s_xc[r * 192 + c] = (__bf16)v;
    }
    __syncthreads();
    { v8f acc = {}; gemm_acc<5>(acc, s_xc, 192, Wpri1, wv, lane); store_relu(s_h1, 128, acc, b_p1, wv, lane); }
    __syncthreads();
    { v8f acc = {}; gemm_acc<4>(acc, s_h1, 128, Wpri2, wv, lane); store_relu(s_h2, 128, acc, b_p2, wv, lane); }
    __syncthreads();
    if (wv < 4) {
      v8f acc = {};
      gemm_acc<4>(acc, s_h2, 128, (wv < 2) ? Wprim : Wpris, wv & 1, lane);
      float* dst = (wv < 2) ? s_mp : s_sp;
      const int colz = (wv & 1) * 16 + nc, lhalf = lane >> 4;
#pragma unroll
      for (int r = 0; r < 8; ++r) {
        float v = acc[r] + hb_p;
        if (wv >= 2) v = softplus_f(v);
        dst[(r + 8 * lhalf) * 32 + colz] = v;
      }
    }
    __syncthreads();

    // ---- latent sample + KL ----
    for (int i = tid; i < 16 * 32; i += 256) {
      int r = i >> 5, zc = i & 31;
      float e  = p.eps[(((size_t)t * kB + (b0 + r)) * kA + a) * kZ + zc];
      float me = s_me[i], se = s_se[i], mp = s_mp[i], sp = s_sp[i];
      float zv = me + se * e;
      s_z[i] = (__bf16)zv;
      float dm = me - mp;
      kl_acc += 0.5f * (2.f * __logf(sp) - 2.f * __logf(se) +
                        (se * se + dm * dm) / (sp * sp) - 1.f);
    }
    __syncthreads();

    // ---- dec input: [y(10) | z(32) | h(128) | pad->192] ----
    for (int i = tid; i < 16 * 192; i += 256) {
      int r = i / 192, c = i - r * 192;
      float v;
      if (c < 10)       v = yrow[(b0 + r) * kY + c];
      else if (c < 42)  v = (float)s_z[r * 32 + (c - 10)];
      else if (c < 170) v = s_hf[r * 128 + (c - 42)];
      else              v = 0.f;
      s_xc[r * 192 + c] = (__bf16)v;
    }
    __syncthreads();
    { v8f acc = {}; gemm_acc<6>(acc, s_xc, 192, Wdec1, wv, lane); store_relu(s_h1, 128, acc, b_d1, wv, lane); }
    __syncthreads();
    { v8f acc = {}; gemm_acc<4>(acc, s_h1, 128, Wdec2, wv, lane); store_relu(s_h2, 128, acc, b_d2, wv, lane); }
    __syncthreads();

    // ---- dec heads (X=2: VALU dots on wave 0) + NLL; others stage GRU input ----
    if (tid < 32) {
      int r = tid >> 1, xi = tid & 1;
      float am = p.dec_mb[a * kX + xi], as = p.dec_sb[a * kX + xi];
#pragma unroll 4
      for (int k = 0; k < kH; ++k) {
        float hv = (float)s_h2[r * 128 + k];
        am += hv * p.dec_mw[(size_t)(a * kH + k) * kX + xi];
        as += hv * p.dec_sw[(size_t)(a * kH + k) * kX + xi];
      }
      float s  = softplus_f(as);
      float xv = xrow[(b0 + r) * kY + a * 2 + xi];
      float d  = (xv - am) / s;
      nll_acc += 0.5f * kLOG2PI + __logf(s) + 0.5f * d * d;
    }
    // GRU input: [x(2) | z(32) | pad->64], stride 64 (reuses s_xc)
    for (int i = tid; i < 16 * 64; i += 256) {
      int r = i >> 6, c = i & 63;
      float v;
      if (c < 2)       v = xrow[(b0 + r) * kY + a * 2 + c];
      else if (c < 34) v = (float)s_z[r * 32 + (c - 2)];
      else             v = 0.f;
      s_xc[r * 64 + c] = (__bf16)v;
    }
    __syncthreads();

    // ---- GRU: all 6 accumulators in registers, no gi/gh LDS round-trip ----
    v8f gir = {}, giz = {}, gin = {}, ghr = {}, ghz = {}, ghn = {};
    gemm_acc<2>(gir, s_xc, 64, Wgih, wv,      lane);
    gemm_acc<2>(giz, s_xc, 64, Wgih, wv + 8,  lane);
    gemm_acc<2>(gin, s_xc, 64, Wgih, wv + 16, lane);
    gemm_acc<4>(ghr, s_hb, 128, Wghh, wv,      lane);
    gemm_acc<4>(ghz, s_hb, 128, Wghh, wv + 8,  lane);
    gemm_acc<4>(ghn, s_hb, 128, Wghh, wv + 16, lane);
    __syncthreads();   // all reads of old s_hb done before updates
    {
      const int lhalf = lane >> 4;
#pragma unroll
      for (int r = 0; r < 8; ++r) {
        int row = r + 8 * lhalf;
        float rg = sigmoid_f(gir[r] + b_ir + ghr[r] + b_hr);
        float ug = sigmoid_f(giz[r] + b_iz + ghz[r] + b_hz);
        float ng = tanhf(gin[r] + b_in + rg * (ghn[r] + b_hn));
        float ho = s_hf[row * 128 + colH];
        float hn = (1.f - ug) * ng + ug * ho;
        s_hf[row * 128 + colH] = hn;
        s_hb[row * 128 + colH] = (__bf16)hn;
      }
    }
    __syncthreads();
  }

  // ---- reduce KL/NLL: wave32 shuffle -> LDS -> global atomic ----
  for (int off = 16; off > 0; off >>= 1) {
    kl_acc  += __shfl_down(kl_acc,  off, 32);
    nll_acc += __shfl_down(nll_acc, off, 32);
  }
  if (lane == 0) { s_red[wv] = kl_acc; s_red[8 + wv] = nll_acc; }
  __syncthreads();
  if (tid == 0) {
    float k = 0.f, n = 0.f;
#pragma unroll
    for (int w = 0; w < 8; ++w) { k += s_red[w]; n += s_red[8 + w]; }
    atomicAdd(&p.out[0], k);
    atomicAdd(&p.out[1], n);
  }
}

// ---------------- host launch ----------------
extern "C" void kernel_launch(void* const* d_in, const int* in_sizes, int n_in,
                              void* d_out, int out_size, void* d_ws, size_t ws_size,
                              hipStream_t stream) {
  (void)in_sizes; (void)n_in; (void)out_size;
  const float* states  = (const float*)d_in[0];
  const float* eps     = (const float*)d_in[1];
  const float* enc_w1  = (const float*)d_in[2];
  const float* enc_b1  = (const float*)d_in[3];
  const float* enc_w2  = (const float*)d_in[4];
  const float* enc_b2  = (const float*)d_in[5];
  const float* enc_mw  = (const float*)d_in[6];
  const float* enc_mb  = (const float*)d_in[7];
  const float* enc_sw  = (const float*)d_in[8];
  const float* enc_sb  = (const float*)d_in[9];
  const float* pri_w1  = (const float*)d_in[10];
  const float* pri_b1  = (const float*)d_in[11];
  const float* pri_w2  = (const float*)d_in[12];
  const float* pri_b2  = (const float*)d_in[13];
  const float* pri_mw  = (const float*)d_in[14];
  const float* pri_mb  = (const float*)d_in[15];
  const float* pri_sw  = (const float*)d_in[16];
  const float* pri_sb  = (const float*)d_in[17];
  const float* dec_w1  = (const float*)d_in[18];
  const float* dec_b1  = (const float*)d_in[19];
  const float* dec_w2  = (const float*)d_in[20];
  const float* dec_b2  = (const float*)d_in[21];
  const float* dec_mw  = (const float*)d_in[22];
  const float* dec_mb  = (const float*)d_in[23];
  const float* dec_sw  = (const float*)d_in[24];
  const float* dec_sb  = (const float*)d_in[25];
  const float* gru_wih = (const float*)d_in[26];
  const float* gru_whh = (const float*)d_in[27];
  const float* gru_bih = (const float*)d_in[28];
  const float* gru_bhh = (const float*)d_in[29];

  __bf16* wsb = (__bf16*)d_ws;   // needs W_TOTAL*2 ~= 2.0 MB (ws_size assumed sufficient)
  (void)ws_size;

  zero_out_kernel<<<1, 32, 0, stream>>>((float*)d_out);

  auto prep = [&](const float* src, size_t off, int K, int Kpad, int N) {
    int total = kA * Kpad * N;
    prep_weights<<<(total + 255) / 256, 256, 0, stream>>>(src, wsb + off, K, Kpad, N);
  };
  prep(enc_w1,  W_ENC_W1,  140, 160, 128);
  prep(enc_w2,  W_ENC_W2,  128, 128, 128);
  prep(enc_mw,  W_ENC_MW,  128, 128, 32);
  prep(enc_sw,  W_ENC_SW,  128, 128, 32);
  prep(pri_w1,  W_PRI_W1,  138, 160, 128);
  prep(pri_w2,  W_PRI_W2,  128, 128, 128);
  prep(pri_mw,  W_PRI_MW,  128, 128, 32);
  prep(pri_sw,  W_PRI_SW,  128, 128, 32);
  prep(dec_w1,  W_DEC_W1,  170, 192, 128);
  prep(dec_w2,  W_DEC_W2,  128, 128, 128);
  prep(gru_wih, W_GRU_WIH,  34,  64, 384);
  prep(gru_whh, W_GRU_WHH, 128, 128, 384);

  P p;
  p.states = states; p.eps = eps;
  p.enc_b1 = enc_b1; p.enc_b2 = enc_b2; p.enc_mb = enc_mb; p.enc_sb = enc_sb;
  p.pri_b1 = pri_b1; p.pri_b2 = pri_b2; p.pri_mb = pri_mb; p.pri_sb = pri_sb;
  p.dec_b1 = dec_b1; p.dec_b2 = dec_b2;
  p.dec_mw = dec_mw; p.dec_mb = dec_mb; p.dec_sw = dec_sw; p.dec_sb = dec_sb;
  p.gru_bih = gru_bih; p.gru_bhh = gru_bhh;
  p.ws = wsb; p.out = (float*)d_out;

  vrnn_kernel<<<dim3(kA * (kB / 16)), dim3(256), 0, stream>>>(p);
}
